// MHSA_13142599925763
// MI455X (gfx1250) — matrix operand order repaired and checked by reading, not compiled, not run on hardware
//
#include <hip/hip_runtime.h>
#include <hip/hip_bf16.h>

// CDNA5 wave32 WMMA types
typedef __attribute__((ext_vector_type(16))) __bf16 v16bf;
typedef __attribute__((ext_vector_type(8)))  float  v8f;

// ---------------------------------------------------------------------------
// ISA 7.12.2 fragment maps (wave32, 16-bit data).
// A 16x32 (MxK): lanes 0-15 hold K={0..7,16..23}, lanes 16-31 hold K={8..15,24..31}
// B 32x16 (KxN): lanes 0-15 hold K={0..7,16..23}, lanes 16-31 hold K={8..15,24..31}, e=8*(K/16)+K%8, lane%16=N
// C/D 16x16 f32: VGPR r holds row M=r (lanes 0-15) / M=r+8 (lanes 16-31), N=lane%16
// Tiles are scattered into LDS already fragment-ordered so each fragment fetch
// is one contiguous 32B ds_load per lane.
// ---------------------------------------------------------------------------
__device__ __forceinline__ void a_map(int row, int k, int& lane, int& e) {
  int hi   = k >> 4;
  int kk   = k & 15;
  int half = kk >> 3;
  int r    = kk & 7;
  int v    = hi * 4 + (r >> 1);
  lane = row + (half << 4);
  e    = (v << 1) + (r & 1);
}

// ---------------------------------------------------------------------------
// 16-lane-row butterfly reductions via DPP16 (VALU only; no LDS round trips).
// XOR1 = quad_perm(1,0,3,2)=0xB1, XOR2 = quad_perm(2,3,0,1)=0x4E,
// row_half_mirror (^7 within 8) = 0x141, row_mirror (^15 within 16) = 0x140.
// The generated group of {1,2,7,15} spans all 16 lanes of a DPP row.
// ---------------------------------------------------------------------------
#if __has_builtin(__builtin_amdgcn_mov_dpp)
#define DPP_XCHG(v, ctrl) \
  __int_as_float(__builtin_amdgcn_mov_dpp(__float_as_int(v), (ctrl), 0xf, 0xf, true))
__device__ __forceinline__ float rowmax16(float v) {
  v = fmaxf(v, DPP_XCHG(v, 0xB1));
  v = fmaxf(v, DPP_XCHG(v, 0x4E));
  v = fmaxf(v, DPP_XCHG(v, 0x141));
  v = fmaxf(v, DPP_XCHG(v, 0x140));
  return v;
}
__device__ __forceinline__ float rowsum16(float v) {
  v += DPP_XCHG(v, 0xB1);
  v += DPP_XCHG(v, 0x4E);
  v += DPP_XCHG(v, 0x141);
  v += DPP_XCHG(v, 0x140);
  return v;
}
#else
__device__ __forceinline__ float rowmax16(float v) {
  for (int off = 8; off >= 1; off >>= 1) v = fmaxf(v, __shfl_xor(v, off, 32));
  return v;
}
__device__ __forceinline__ float rowsum16(float v) {
  for (int off = 8; off >= 1; off >>= 1) v += __shfl_xor(v, off, 32);
  return v;
}
#endif

// ===========================================================================
// Generic 128x128-tile GEMM: C[MxN] = A[MxK] * B[KxN]
// 256 threads = 8 waves (4 along M x 2 along N), wave tile 32x64 = 8 WMMA accs.
// Double-buffered LDS (single barrier per K-step); global loads for step k+1
// are issued before the compute of step k; prefetch covers step k+2.
// ===========================================================================
template<bool A_F32, bool B_F32, bool OUT_F32>
__global__ __launch_bounds__(256)
void gemm_wmma_k(const void* __restrict__ Ap, const void* __restrict__ Bp,
                 void* __restrict__ Cp, int M, int N, int K)
{
  __shared__ __align__(64) __bf16 ldsA[2][8][512];   // ping-pong, 8 M-subtiles of 16x32
  __shared__ __align__(64) __bf16 ldsB[2][8][512];   // ping-pong, 8 N-subtiles of 32x16
  (void)M;

  const int tid   = threadIdx.x;
  const int lane  = tid & 31;
  const int wid   = tid >> 5;
  const int waveM = wid & 3;
  const int waveN = wid >> 2;
  const long long blockM = (long long)blockIdx.y * 128;
  const long long blockN = (long long)blockIdx.x * 128;

  const float*  Af = (const float*)Ap;
  const __bf16* Ab = (const __bf16*)Ap;
  const float*  Bf = (const float*)Bp;
  const __bf16* Bb = (const __bf16*)Bp;

  v8f acc[2][4];
  #pragma unroll
  for (int s = 0; s < 2; ++s)
    #pragma unroll
    for (int nt = 0; nt < 4; ++nt)
      #pragma unroll
      for (int r = 0; r < 8; ++r) acc[s][nt][r] = 0.0f;

  // per-thread staging geometry (16 contiguous elements each for A and B)
  const int rowL   = tid >> 1;
  const int kL0    = (tid & 1) << 4;
  const int aSub   = rowL >> 4;
  const int aRowIn = rowL & 15;
  const int krow   = tid >> 3;
  const int nL0    = (tid & 7) << 4;
  const int bSub   = tid & 7;
  const int nK     = K >> 5;

  auto loadRegs = [&](int kb, __bf16* av, __bf16* bv) {
    const int k0 = kb << 5;
    if (A_F32) {
      const float4* p = (const float4*)(Af + (blockM + rowL) * (long long)K + k0 + kL0);
      #pragma unroll
      for (int q = 0; q < 4; ++q) {
        float4 f = p[q];
        av[q*4+0] = (__bf16)f.x; av[q*4+1] = (__bf16)f.y;
        av[q*4+2] = (__bf16)f.z; av[q*4+3] = (__bf16)f.w;
      }
    } else {
      v16bf t = *(const v16bf*)(Ab + (blockM + rowL) * (long long)K + k0 + kL0);
      #pragma unroll
      for (int q = 0; q < 16; ++q) av[q] = t[q];
    }
    if (B_F32) {
      const float4* p = (const float4*)(Bf + (long long)(k0 + krow) * N + blockN + nL0);
      #pragma unroll
      for (int q = 0; q < 4; ++q) {
        float4 f = p[q];
        bv[q*4+0] = (__bf16)f.x; bv[q*4+1] = (__bf16)f.y;
        bv[q*4+2] = (__bf16)f.z; bv[q*4+3] = (__bf16)f.w;
      }
    } else {
      v16bf t = *(const v16bf*)(Bb + (long long)(k0 + krow) * N + blockN + nL0);
      #pragma unroll
      for (int q = 0; q < 16; ++q) bv[q] = t[q];
    }
    // global_prefetch for the tile after next (register buffer covers +1)
    if (kb + 1 < nK) {
      const int kp = (kb + 1) << 5;
      __builtin_prefetch((const char*)Ap +
          ((blockM + rowL) * (long long)K + kp + kL0) * (A_F32 ? 4 : 2), 0, 0);
      __builtin_prefetch((const char*)Bp +
          ((long long)(kp + krow) * N + blockN + nL0) * (B_F32 ? 4 : 2), 0, 0);
    }
  };

  auto scatter = [&](const __bf16* av, const __bf16* bv, int buf) {
    #pragma unroll
    for (int q = 0; q < 16; ++q) {
      int ln, e; a_map(aRowIn, kL0 + q, ln, e);
      ldsA[buf][aSub][ln * 16 + e] = av[q];
    }
    const int bl = ((krow >> 3) & 1) << 4;
    const int be = ((krow >> 4) << 3) + (krow & 7);
    #pragma unroll
    for (int q = 0; q < 16; ++q)
      ldsB[buf][bSub][(q + bl) * 16 + be] = bv[q];
  };

  auto compute = [&](int buf) {
    v16bf afrag[2];
    #pragma unroll
    for (int s = 0; s < 2; ++s)
      afrag[s] = *(const v16bf*)&ldsA[buf][waveM * 2 + s][lane * 16];
    #pragma unroll
    for (int nt = 0; nt < 4; ++nt) {
      v16bf bfrag = *(const v16bf*)&ldsB[buf][waveN * 4 + nt][lane * 16];
      #pragma unroll
      for (int s = 0; s < 2; ++s)
        acc[s][nt] = __builtin_amdgcn_wmma_f32_16x16x32_bf16(
            false, afrag[s], false, bfrag, (short)0, acc[s][nt], false, false);
    }
  };

  // prologue: stage tile 0
  {
    __bf16 av[16], bv[16];
    loadRegs(0, av, bv);
    scatter(av, bv, 0);
  }
  __syncthreads();

  int cur = 0;
  for (int kb = 1; kb < nK; ++kb) {
    __bf16 av[16], bv[16];
    loadRegs(kb, av, bv);        // global loads in flight during compute
    compute(cur);                // reads lds[cur]
    scatter(av, bv, cur ^ 1);    // writes lds[cur^1] (no reader this iter)
    __syncthreads();             // separates buffer-role swap across iterations
    cur ^= 1;
  }
  compute(cur);

  // epilogue (C/D layout)
  const int half = lane >> 4, ln = lane & 15;
  #pragma unroll
  for (int s = 0; s < 2; ++s)
    #pragma unroll
    for (int nt = 0; nt < 4; ++nt)
      #pragma unroll
      for (int r = 0; r < 8; ++r) {
        long long m  = blockM + waveM * 32 + s * 16 + r + half * 8;
        long long nn = blockN + waveN * 64 + nt * 16 + ln;
        if (OUT_F32) ((float*)Cp)[m * N + nn] = acc[s][nt][r];
        else         ((__bf16*)Cp)[m * N + nn] = (__bf16)acc[s][nt][r];
      }
}

// ===========================================================================
// Flash attention over one (head, segment, 128-row q-block).
// 8 waves; wave w owns q rows [16w,16w+16). 64-key blocks streamed with online
// softmax; K/V LDS double-buffered; softmax scale folded (exactly) into bf16 Q.
// ===========================================================================
__global__ __launch_bounds__(256)
void attn_fmha_k(const __bf16* __restrict__ qkv, __bf16* __restrict__ y)
{
  constexpr int Cdim = 1024, TC = 3072, HD = 64, SEG = 512;
  __shared__ __align__(64) __bf16 ldsQ[8][2][512];     // per-wave Q A-frags
  __shared__ __align__(64) __bf16 ldsK[2][4][2][512];  // ping-pong K^T B-frags
  __shared__ __align__(64) __bf16 ldsV[2][4][2][512];  // ping-pong V B-frags
  __shared__ __align__(64) __bf16 ldsP[8][2][512];     // per-wave P A-frags

  const int tid  = threadIdx.x;
  const int lane = tid & 31;
  const int wid  = tid >> 5;
  const int h    = blockIdx.x >> 5;
  const int seg  = blockIdx.x & 31;
  const int qb   = blockIdx.y;
  const long long segBase = (long long)seg * SEG;
  const int qoff = h * HD;

  // ---- stage Q (128x64), pre-scaled by 1/sqrt(HD)=0.125 (exact in bf16) ----
  {
    const int rowL = tid >> 1;
    const int d0   = (tid & 1) << 5;
    const int ks   = tid & 1;
    long long t = segBase + (long long)qb * 128 + rowL;
    const v16bf* p = (const v16bf*)(qkv + t * TC + qoff + d0);
    v16bf c0 = p[0], c1 = p[1];
    const int sub = rowL >> 4, rIn = rowL & 15;
    #pragma unroll
    for (int j = 0; j < 16; ++j) {
      int ln, e; a_map(rIn, j, ln, e);
      ldsQ[sub][ks][ln * 16 + e] = (__bf16)((float)c0[j] * 0.125f);
    }
    #pragma unroll
    for (int j = 0; j < 16; ++j) {
      int ln, e; a_map(rIn, 16 + j, ln, e);
      ldsQ[sub][ks][ln * 16 + e] = (__bf16)((float)c1[j] * 0.125f);
    }
  }

  float m_i[8], l_i[8];
  v8f accO[4];
  #pragma unroll
  for (int r = 0; r < 8; ++r) { m_i[r] = -3.0e38f; l_i[r] = 0.0f; }
  #pragma unroll
  for (int nt = 0; nt < 4; ++nt)
    #pragma unroll
    for (int r = 0; r < 8; ++r) accO[nt][r] = 0.0f;

  // K/V staging geometry: thread covers key-row kr, d in [d0, d0+16)
  const int kr   = tid >> 2;
  const int d0   = (tid & 3) << 4;
  const int ksD  = d0 >> 5;
  const int subN = kr >> 4, nIn = kr & 15;
  const int ksV  = kr >> 5;
  const int lV   = ((kr >> 3) & 1) << 4;
  const int eV   = (((kr & 31) >> 4) << 3) + (kr & 7);
  const int subV = d0 >> 4;

  auto loadKV = [&](int kb, v16bf& kc, v16bf& vc) {
    long long t = segBase + (long long)kb * 64 + kr;
    kc = *(const v16bf*)(qkv + t * TC +     Cdim + qoff + d0);
    vc = *(const v16bf*)(qkv + t * TC + 2 * Cdim + qoff + d0);
    if (kb + 1 < 8) {
      long long tp = segBase + (long long)(kb + 1) * 64 + kr;
      __builtin_prefetch((const char*)(qkv + tp * TC +     Cdim + qoff + d0), 0, 0);
      __builtin_prefetch((const char*)(qkv + tp * TC + 2 * Cdim + qoff + d0), 0, 0);
    }
  };
  auto scatterKV = [&](const v16bf& kc, const v16bf& vc, int buf) {
    #pragma unroll
    for (int j = 0; j < 16; ++j) {
      int kk = (d0 & 31) + j;
      ldsK[buf][subN][ksD][(nIn + (((kk >> 3) & 1) << 4)) * 16 + ((kk >> 4) << 3) + (kk & 7)] = kc[j];
      ldsV[buf][subV][ksV][(j + lV) * 16 + eV] = vc[j];
    }
  };

  // prologue: stage KV block 0
  {
    v16bf kc, vc;
    loadKV(0, kc, vc);
    scatterKV(kc, vc, 0);
  }
  __syncthreads();

  int cur = 0;
  for (int kb = 0; kb < 8; ++kb) {
    v16bf kcN, vcN;
    const bool have = (kb + 1) < 8;
    if (have) loadKV(kb + 1, kcN, vcN);   // next block in flight during compute

    // ---- S = Q K^T (scale pre-folded into Q) ----
    v8f s[4];
    #pragma unroll
    for (int nt = 0; nt < 4; ++nt)
      #pragma unroll
      for (int r = 0; r < 8; ++r) s[nt][r] = 0.0f;
    #pragma unroll
    for (int ks = 0; ks < 2; ++ks) {
      v16bf aq = *(const v16bf*)&ldsQ[wid][ks][lane * 16];
      #pragma unroll
      for (int nt = 0; nt < 4; ++nt) {
        v16bf bk = *(const v16bf*)&ldsK[cur][nt][ks][lane * 16];
        s[nt] = __builtin_amdgcn_wmma_f32_16x16x32_bf16(
            false, aq, false, bk, (short)0, s[nt], false, false);
      }
    }

    // ---- online softmax (DPP16 butterfly within each 16-lane row) ----
    float alpha[8], rs[8];
    #pragma unroll
    for (int r = 0; r < 8; ++r) {
      float rm = rowmax16(fmaxf(fmaxf(s[0][r], s[1][r]), fmaxf(s[2][r], s[3][r])));
      float mn = fmaxf(m_i[r], rm);
      alpha[r] = __expf(m_i[r] - mn);
      m_i[r] = mn;
      rs[r] = 0.0f;
    }
    #pragma unroll
    for (int nt = 0; nt < 4; ++nt)
      #pragma unroll
      for (int r = 0; r < 8; ++r) {
        float p = __expf(s[nt][r] - m_i[r]);
        s[nt][r] = p;
        rs[r] += p;
      }
    #pragma unroll
    for (int r = 0; r < 8; ++r) {
      l_i[r] = l_i[r] * alpha[r] + rowsum16(rs[r]);
    }
    #pragma unroll
    for (int nt = 0; nt < 4; ++nt)
      #pragma unroll
      for (int r = 0; r < 8; ++r) accO[nt][r] *= alpha[r];

    // ---- transpose P (C-layout -> A-fragment) through per-wave LDS ----
    {
      const int halfL = lane >> 4, ln = lane & 15;
      #pragma unroll
      for (int nt = 0; nt < 4; ++nt) {
        int ksP = nt >> 1;
        int kk  = ((nt & 1) << 4) + ln;
        #pragma unroll
        for (int r = 0; r < 8; ++r) {
          int l2, e2; a_map(r + (halfL << 3), kk, l2, e2);
          ldsP[wid][ksP][l2 * 16 + e2] = (__bf16)s[nt][r];
        }
      }
    }
    // per-wave region: same-wave DS ops are in-order; no barrier needed

    // ---- O += P V ----
    #pragma unroll
    for (int ks = 0; ks < 2; ++ks) {
      v16bf ap = *(const v16bf*)&ldsP[wid][ks][lane * 16];
      #pragma unroll
      for (int nt = 0; nt < 4; ++nt) {
        v16bf bv = *(const v16bf*)&ldsV[cur][nt][ks][lane * 16];
        accO[nt] = __builtin_amdgcn_wmma_f32_16x16x32_bf16(
            false, ap, false, bv, (short)0, accO[nt], false, false);
      }
    }

    if (have) scatterKV(kcN, vcN, cur ^ 1);  // fill other buffer (no reader now)
    __syncthreads();                          // swap buffer roles
    cur ^= 1;
  }

  // ---- epilogue: y = O / l ----
  const int half = lane >> 4, ln = lane & 15;
  float inv[8];
  #pragma unroll
  for (int r = 0; r < 8; ++r) inv[r] = 1.0f / l_i[r];
  #pragma unroll
  for (int nt = 0; nt < 4; ++nt)
    #pragma unroll
    for (int r = 0; r < 8; ++r) {
      long long t = segBase + (long long)qb * 128 + wid * 16 + r + (half << 3);
      y[t * Cdim + qoff + nt * 16 + ln] = (__bf16)(accO[nt][r] * inv[r]);
    }
}

// ===========================================================================
extern "C" void kernel_launch(void* const* d_in, const int* in_sizes, int n_in,
                              void* d_out, int out_size, void* d_ws, size_t ws_size,
                              hipStream_t stream) {
  (void)in_sizes; (void)n_in; (void)out_size; (void)ws_size;
  constexpr int T = 16384, C = 1024;
  const float* x      = (const float*)d_in[0];
  const float* w_attn = (const float*)d_in[1];
  const float* w_proj = (const float*)d_in[2];
  // d_in[3] = split_sections: segments are equal (T/32), layout handled directly.
  float* out = (float*)d_out;

  __bf16* qkv_bf = (__bf16*)d_ws;                  // [T, 3C] bf16 (96 MB)
  __bf16* y_bf   = qkv_bf + (size_t)T * 3 * C;     // [T, C]  bf16 (32 MB)

  // 1) qkv = x @ w_attn  (f32 in, bf16 out)
  gemm_wmma_k<true, true, false>
      <<<dim3(3 * C / 128, T / 128), 256, 0, stream>>>(x, w_attn, qkv_bf, T, 3 * C, C);
  // 2) y = segment-blocked flash attention
  attn_fmha_k<<<dim3(16 * 32, 4), 256, 0, stream>>>(qkv_bf, y_bf);
  // 3) out = y @ w_proj  (bf16 A, f32 B, f32 out)
  gemm_wmma_k<false, true, true>
      <<<dim3(C / 128, T / 128), 256, 0, stream>>>(y_bf, w_proj, out, T, C, C);
}
